// LoopVisibleLSTM_63170378989791
// MI455X (gfx1250) — matrix operand order repaired and checked by reading, not compiled, not run on hardware
//
#include <hip/hip_runtime.h>
#include <hip/hip_bf16.h>

// ---------------- problem constants ----------------
#define TT   2048
#define BB   32
#define DIN  256
#define HH   256
#define LL   2
#define G4H  1024          // 4*H
#define TBR  (TT*BB)       // 65536 rows of the big GEMMs
#define NBLK 16            // recurrent blocks (one H/16 column slice each)
#define WPAD 272           // padded LDS row (bf16 elems): 544B rows -> bank-conflict-free-ish

typedef __attribute__((ext_vector_type(16))) __bf16 v16bf;
typedef __attribute__((ext_vector_type(8)))  __bf16 v8bf;
typedef __attribute__((ext_vector_type(8)))  float  v8f;

// ---------------- WMMA fragment loaders (CDNA5 wave32 layouts) ----------------
// A: 16x32 bf16, row-major source, leading dim ld.
__device__ __forceinline__ v16bf load_a16x32(const __bf16* __restrict__ base,
                                             int ld, int k0, int lane) {
  const int row  = lane & 15;
  const int koff = (lane < 16) ? 0 : 16;
  const __bf16* p = base + row * ld + k0 + koff;
  v8bf lo = *(const v8bf*)(p);        // K = koff .. koff+7
  v8bf hi = *(const v8bf*)(p + 8);    // K = koff+8 .. koff+15
  return __builtin_shufflevector(lo, hi, 0,1,2,3,4,5,6,7,8,9,10,11,12,13,14,15);
}

// B: 32x16 bf16 where B = W^T (W row-major [16 rows of N, ld=K]).
__device__ __forceinline__ v16bf load_bT32x16(const __bf16* __restrict__ Wbase,
                                              int ld, int k0, int lane) {
  const int colN = lane & 15;
  const int koff = (lane < 16) ? 0 : 16;
  return *(const v16bf*)(Wbase + colN * ld + k0 + koff);
}

// fast gate nonlinearities: v_exp_f32 + v_rcp_f32, no IEEE division sequences
__device__ __forceinline__ float sigmoidf_(float x) {
  return __builtin_amdgcn_rcpf(1.f + __expf(-x));
}
__device__ __forceinline__ float tanhf_(float x) {
  return 2.f * __builtin_amdgcn_rcpf(1.f + __expf(-2.f * x)) - 1.f;
}

// ---------------- small utility kernels ----------------
__global__ void k_f32_to_bf16(const float* __restrict__ src, __bf16* __restrict__ dst, int n) {
  int i = blockIdx.x * blockDim.x + threadIdx.x;
  if (i < n) dst[i] = (__bf16)src[i];
}

__global__ void k_bias_combine(const float* __restrict__ bih, const float* __restrict__ bhh,
                               float* __restrict__ bias, int n) {
  int i = blockIdx.x * blockDim.x + threadIdx.x;
  if (i < n) bias[i] = bih[i] + bhh[i];
}

// init: barrier counters = 0, h ping-pong buffers[0] = initial hidden states (bf16)
__global__ void k_init(const float* __restrict__ h_init, __bf16* __restrict__ hbuf0,
                       __bf16* __restrict__ hbuf1, unsigned* __restrict__ bar) {
  int i = blockIdx.x * blockDim.x + threadIdx.x;   // 8192 threads
  hbuf0[i] = (__bf16)h_init[i];
  hbuf1[i] = (__bf16)h_init[BB * HH + i];
  if (i < 2) bar[i] = 0u;
}

// broadcast final top-layer hidden into the "backward" half of the output
__global__ void k_broadcast_bwd(const float* __restrict__ hfin, float* __restrict__ out, int total) {
  int i = blockIdx.x * blockDim.x + threadIdx.x;
  if (i >= total) return;
  int u = i & 255, b = (i >> 8) & 31, t = i >> 13;
  out[(size_t)t * (BB * 2 * HH) + b * (2 * HH) + HH + u] = hfin[b * HH + u];
}

// ---------------- T-parallel precompute GEMM: C = A * W^T + bias ----------------
template <typename OutT>
__global__ void k_gemm_bf16(const __bf16* __restrict__ A, const __bf16* __restrict__ W,
                            const float* __restrict__ bias, OutT* __restrict__ C,
                            int M, int N, int K) {
  const int lane = threadIdx.x & 31;
  const int wave = threadIdx.x >> 5;
  const int ntiles = N >> 4;
  const int tile = blockIdx.x * (blockDim.x >> 5) + wave;
  if (tile >= (M >> 4) * ntiles) return;           // wave-uniform exit: EXEC stays all-1s
  const int mt = tile / ntiles, nt = tile % ntiles;
  const int Mb  = (lane < 16) ? 0 : 8;
  const int col = lane & 15;

  v8f acc;
  const float bb = bias[nt * 16 + col];
#pragma unroll
  for (int r = 0; r < 8; ++r) acc[r] = bb;

  const __bf16* Ab = A + (size_t)mt * 16 * K;
  const __bf16* Wb = W + (size_t)nt * 16 * K;
  for (int kk = 0; kk < K; kk += 32) {
    v16bf a = load_a16x32(Ab, K, kk, lane);
    v16bf b = load_bT32x16(Wb, K, kk, lane);
    acc = __builtin_amdgcn_wmma_f32_16x16x32_bf16(false, a, false, b, (short)0, acc, false, false);
  }
  OutT* Cb = C + (size_t)mt * 16 * N + nt * 16;
#pragma unroll
  for (int r = 0; r < 8; ++r) Cb[(size_t)(Mb + r) * N + col] = (OutT)acc[r];
}

// ---------------- device-scope barrier (L2 atomics, self-resetting) ----------------
__device__ __forceinline__ void gbarrier(unsigned* cnt, unsigned* phase, unsigned expected) {
  __syncthreads();
  if (threadIdx.x == 0) {
    __threadfence();
    unsigned prev = atomicAdd(cnt, 1u);
    if (prev == NBLK - 1u) {
      atomicExch(cnt, 0u);
      __threadfence();
      atomicAdd(phase, 1u);
    } else {
      while (__hip_atomic_load(phase, __ATOMIC_RELAXED, __HIP_MEMORY_SCOPE_AGENT) < expected)
        __builtin_amdgcn_s_sleep(1);
    }
    __threadfence();
  }
  __syncthreads();
}

// ---------------- distributed serial LSTM scan: 16 WGPs, LDS-resident weights ------
// Block bx owns the 16-wide hidden-unit slice u in [bx*16, bx*16+16) of BOTH layers.
// Its weight rows (i,f,g,o blocks of Whh0 / Wih1 / Whh1) are staged into LDS once
// (3 x 64 rows x 256 K, rows padded to WPAD) and feed every WMMA via ds_load.
// 8 waves/block: wave w computes tile (mt = w>>2, gate = w&3) of a [32 x 64] slice.
// c-state lives in registers; h ping-pong buffers live in global (L2-resident, 16KB).
__global__ void k_recurrent16(const __bf16* __restrict__ Wih_bf,  // [L,4H,H]
                              const __bf16* __restrict__ Whh_bf,  // [L,4H,H]
                              const float*  __restrict__ bias,    // [L,4H] bih+bhh
                              const float*  __restrict__ pre0,    // [T*B,4H]
                              const float*  __restrict__ c_init,  // [L,B,H]
                              __bf16*       __restrict__ hbuf0,   // [2][B*H] ping-pong
                              __bf16*       __restrict__ hbuf1,   // [2][B*H]
                              float*        __restrict__ out,     // [T,B,2H]
                              float*        __restrict__ hfin,    // [B,H]
                              unsigned*     __restrict__ bar) {   // [2] cnt, phase
  extern __shared__ char smem[];
  __bf16* w0  = (__bf16*)smem;            // Whh0 slice: [4 gates][16 rows][WPAD]
  __bf16* wi1 = w0  + 64 * WPAD;          // Wih1 slice
  __bf16* wh1 = wi1 + 64 * WPAD;          // Whh1 slice
  float*  gates = (float*)(wh1 + 64 * WPAD);  // [2 mt][4 gate][16 r][16 c] = 8KB

  const int bx   = blockIdx.x;            // owned u-slice
  const int tid  = threadIdx.x;
  const int lane = tid & 31;
  const int wave = tid >> 5;              // 0..7
  const int mtw  = wave >> 2;             // 0..1
  const int gw   = wave & 3;              // gate of this wave's tile
  const int Mb   = (lane < 16) ? 0 : 8;
  const int col  = lane & 15;

  // ---- stage weight slices into LDS (once) ----
  const __bf16* Wsrc[3] = { Whh_bf, Wih_bf + (size_t)G4H * HH, Whh_bf + (size_t)G4H * HH };
  __bf16* Wdst[3] = { w0, wi1, wh1 };
#pragma unroll 1
  for (int m = 0; m < 3; ++m) {
#pragma unroll 1
    for (int j = 0; j < 8; ++j) {                 // 2048 v8bf chunks / 256 threads
      int q  = j * 256 + tid;
      int rl = q >> 5;                            // local row 0..63  (g*16 + rr)
      int kc = (q & 31) * 8;
      int g  = rl >> 4, rr = rl & 15;
      v8bf v = *(const v8bf*)(Wsrc[m] + ((size_t)(g * 256 + bx * 16 + rr) * HH + kc));
      *(v8bf*)(Wdst[m] + rl * WPAD + kc) = v;
    }
  }

  // ---- register-resident c-state: thread owns e = k*256+tid -> (b,u) ----
  float c0r[2], c1r[2];
#pragma unroll
  for (int k = 0; k < 2; ++k) {
    int e = k * 256 + tid;                        // 0..511 within slice
    int b = e >> 4, cc = e & 15;
    c0r[k] = c_init[b * HH + bx * 16 + cc];
    c1r[k] = c_init[BB * HH + b * HH + bx * 16 + cc];
  }
  __syncthreads();

  const float* bias1 = bias + G4H;
  unsigned barphase = 0;

#pragma unroll 1
  for (int t = 0; t < TT; ++t) {
    const int p = t & 1;
    const __bf16* h0r = hbuf0 + p * (BB * HH);
    __bf16*       h0w = hbuf0 + (1 - p) * (BB * HH);
    const __bf16* h1r = hbuf1 + p * (BB * HH);
    __bf16*       h1w = hbuf1 + (1 - p) * (BB * HH);
    const float* pre_t = pre0 + (size_t)t * BB * G4H;
    if (t + 1 < TT)
      __builtin_prefetch(pre0 + (size_t)(t + 1) * BB * G4H + wave * G4H, 0, 1);

    // ---- layer 0: acc = pre0[t] + h0 @ Whh0^T  (this wave: tile (mtw, gw)) ----
    {
      v8f acc;
#pragma unroll
      for (int r = 0; r < 8; ++r)
        acc[r] = pre_t[(size_t)(mtw * 16 + Mb + r) * G4H + gw * 256 + bx * 16 + col];
#pragma unroll
      for (int kk = 0; kk < 8; ++kk) {
        v16bf a = load_a16x32(h0r + mtw * 16 * HH, HH, kk * 32, lane);     // global (L2)
        v16bf b = load_bT32x16(w0 + gw * 16 * WPAD, WPAD, kk * 32, lane);  // ds_load
        acc = __builtin_amdgcn_wmma_f32_16x16x32_bf16(false, a, false, b, (short)0, acc, false, false);
      }
#pragma unroll
      for (int r = 0; r < 8; ++r)
        gates[((mtw * 4 + gw) * 16 + Mb + r) * 16 + col] = acc[r];
    }
    __syncthreads();

    // ---- layer 0 gate combine (i,f,g,o), write h0_new slice ----
#pragma unroll
    for (int k = 0; k < 2; ++k) {
      int e = k * 256 + tid;
      int b = e >> 4, cc = e & 15;
      int mt = b >> 4, r = b & 15;
      float gi = gates[((mt * 4 + 0) * 16 + r) * 16 + cc];
      float gf = gates[((mt * 4 + 1) * 16 + r) * 16 + cc];
      float gg = gates[((mt * 4 + 2) * 16 + r) * 16 + cc];
      float go = gates[((mt * 4 + 3) * 16 + r) * 16 + cc];
      float cn = sigmoidf_(gf) * c0r[k] + sigmoidf_(gi) * tanhf_(gg);
      c0r[k] = cn;
      h0w[b * HH + bx * 16 + cc] = (__bf16)(sigmoidf_(go) * tanhf_(cn));
    }

    // ---- all blocks exchange h0_new (single device barrier per step) ----
    gbarrier(bar, bar + 1, ++barphase);

    // ---- layer 1: acc = bias1 + h0_new @ Wih1^T + h1 @ Whh1^T ----
    {
      v8f acc;
      const float bb = bias1[gw * 256 + bx * 16 + col];
#pragma unroll
      for (int r = 0; r < 8; ++r) acc[r] = bb;
#pragma unroll
      for (int kk = 0; kk < 8; ++kk) {
        v16bf a = load_a16x32(h0w + mtw * 16 * HH, HH, kk * 32, lane);
        v16bf b = load_bT32x16(wi1 + gw * 16 * WPAD, WPAD, kk * 32, lane);
        acc = __builtin_amdgcn_wmma_f32_16x16x32_bf16(false, a, false, b, (short)0, acc, false, false);
      }
#pragma unroll
      for (int kk = 0; kk < 8; ++kk) {
        v16bf a = load_a16x32(h1r + mtw * 16 * HH, HH, kk * 32, lane);
        v16bf b = load_bT32x16(wh1 + gw * 16 * WPAD, WPAD, kk * 32, lane);
        acc = __builtin_amdgcn_wmma_f32_16x16x32_bf16(false, a, false, b, (short)0, acc, false, false);
      }
#pragma unroll
      for (int r = 0; r < 8; ++r)
        gates[((mtw * 4 + gw) * 16 + Mb + r) * 16 + col] = acc[r];
    }
    __syncthreads();

    // ---- layer 1 gate combine, emit forward output slice ----
    float* out_t = out + (size_t)t * BB * (2 * HH);
#pragma unroll
    for (int k = 0; k < 2; ++k) {
      int e = k * 256 + tid;
      int b = e >> 4, cc = e & 15;
      int mt = b >> 4, r = b & 15;
      float gi = gates[((mt * 4 + 0) * 16 + r) * 16 + cc];
      float gf = gates[((mt * 4 + 1) * 16 + r) * 16 + cc];
      float gg = gates[((mt * 4 + 2) * 16 + r) * 16 + cc];
      float go = gates[((mt * 4 + 3) * 16 + r) * 16 + cc];
      float cn = sigmoidf_(gf) * c1r[k] + sigmoidf_(gi) * tanhf_(gg);
      c1r[k] = cn;
      float hn = sigmoidf_(go) * tanhf_(cn);
      h1w[b * HH + bx * 16 + cc] = (__bf16)hn;
      out_t[b * (2 * HH) + bx * 16 + cc] = hn;
      if (t == TT - 1) hfin[b * HH + bx * 16 + cc] = hn;
    }
    __syncthreads();   // gates reuse next iteration; h1w ordered by next step's barrier
  }
}

// ---------------- host launcher ----------------
extern "C" void kernel_launch(void* const* d_in, const int* in_sizes, int n_in,
                              void* d_out, int out_size, void* d_ws, size_t ws_size,
                              hipStream_t stream) {
  const float* in_f    = (const float*)d_in[0];
  const float* Winit_f = (const float*)d_in[1];
  const float* b_init  = (const float*)d_in[2];
  const float* Wih_f   = (const float*)d_in[3];
  const float* Whh_f   = (const float*)d_in[4];
  const float* bih     = (const float*)d_in[5];
  const float* bhh     = (const float*)d_in[6];
  const float* h0      = (const float*)d_in[7];
  const float* c0      = (const float*)d_in[8];
  float* out = (float*)d_out;

  char* ws = (char*)d_ws;
  size_t off = 0;
  auto carve = [&](size_t bytes) { char* p = ws + off; off = (off + bytes + 255) & ~(size_t)255; return p; };
  __bf16* in_bf    = (__bf16*)carve((size_t)TBR * DIN * 2);
  __bf16* winit_bf = (__bf16*)carve((size_t)HH * DIN * 2);
  __bf16* wih_bf   = (__bf16*)carve((size_t)LL * G4H * HH * 2);
  __bf16* whh_bf   = (__bf16*)carve((size_t)LL * G4H * HH * 2);
  float*  bias     = (float*) carve((size_t)LL * G4H * 4);
  __bf16* xinit_bf = (__bf16*)carve((size_t)TBR * HH * 2);
  float*  pre0     = (float*) carve((size_t)TBR * G4H * 4);
  __bf16* hbuf0    = (__bf16*)carve((size_t)2 * BB * HH * 2);
  __bf16* hbuf1    = (__bf16*)carve((size_t)2 * BB * HH * 2);
  float*  hfin     = (float*) carve((size_t)BB * HH * 4);
  unsigned* bar    = (unsigned*)carve(256);
  (void)ws_size; (void)n_in; (void)in_sizes; (void)out_size;

  // 1) bf16 conversions + combined biases
  k_f32_to_bf16<<<(TBR * DIN) / 256, 256, 0, stream>>>(in_f, in_bf, TBR * DIN);
  k_f32_to_bf16<<<(HH * DIN) / 256, 256, 0, stream>>>(Winit_f, winit_bf, HH * DIN);
  k_f32_to_bf16<<<(LL * G4H * HH) / 256, 256, 0, stream>>>(Wih_f, wih_bf, LL * G4H * HH);
  k_f32_to_bf16<<<(LL * G4H * HH) / 256, 256, 0, stream>>>(Whh_f, whh_bf, LL * G4H * HH);
  k_bias_combine<<<(LL * G4H) / 256, 256, 0, stream>>>(bih, bhh, bias, LL * G4H);

  // 2) T-parallel precompute (compute-bound, pure WMMA)
  k_gemm_bf16<__bf16><<<(TBR / 16) * (HH / 16) / 8, 256, 0, stream>>>(
      in_bf, winit_bf, b_init, xinit_bf, TBR, HH, DIN);
  k_gemm_bf16<float><<<(TBR / 16) * (G4H / 16) / 8, 256, 0, stream>>>(
      xinit_bf, wih_bf, bias, pre0, TBR, G4H, HH);

  // 3) init barrier state + h ping-pong buffers (ws is poisoned; runs every call)
  k_init<<<(BB * HH) / 256, 256, 0, stream>>>(h0, hbuf0, hbuf1, bar);

  // 4) distributed serial scan: 16 blocks, LDS-resident weight slices (~110KB/WGP)
  const int lds_bytes = 3 * 64 * WPAD * 2 + 2 * 4 * 16 * 16 * 4;
  hipFuncSetAttribute(reinterpret_cast<const void*>(k_recurrent16),
                      hipFuncAttributeMaxDynamicSharedMemorySize, lds_bytes);
  k_recurrent16<<<NBLK, 256, lds_bytes, stream>>>(wih_bf, whh_bf, bias, pre0, c0,
                                                  hbuf0, hbuf1, out, hfin, bar);

  // 5) "backward" half = final top-layer hidden broadcast over T
  k_broadcast_bwd<<<(TT * BB * HH) / 256, 256, 0, stream>>>(hfin, out, TT * BB * HH);
}
